// AttentionLayer_50371376447725
// MI455X (gfx1250) — matrix operand order repaired and checked by reading, not compiled, not run on hardware
//
#include <hip/hip_runtime.h>

typedef __attribute__((ext_vector_type(16))) __bf16 v16bf;
typedef __attribute__((ext_vector_type(8)))  float  v8f;

#define B_   2048
#define T_   200
#define E_   128
#define H1_  64
#define H2_  32
#define MT_  13                      // ceil(200/16) M-tiles of 16 rows
#define NEG_INF_F (-4294967295.0f)   // -(2^32)+1, matches reference

__device__ __forceinline__ unsigned int packbf2(float a, float b) {
  unsigned int ua = __float_as_uint(a);
  unsigned int ub = __float_as_uint(b);
  unsigned int ra = (ua + 0x7FFFu + ((ua >> 16) & 1u)) >> 16;   // RNE f32->bf16
  unsigned int rb = (ub + 0x7FFFu + ((ub >> 16) & 1u)) >> 16;
  return (ra & 0xFFFFu) | (rb << 16);
}
__device__ __forceinline__ unsigned short f2bf(float a) {
  unsigned int ua = __float_as_uint(a);
  return (unsigned short)((ua + 0x7FFFu + ((ua >> 16) & 1u)) >> 16);
}
__device__ __forceinline__ float bf2f(unsigned int hbits) {
  return __uint_as_float(hbits << 16);
}

__global__ __launch_bounds__(256, 4)   // cap VGPRs <= 256: no vgpr-msb churn, 2 WGs/WGP
void din_attn_kernel(const float* __restrict__ queries, const float* __restrict__ keys,
                     const int* __restrict__ keys_id,
                     const float* __restrict__ W1, const float* __restrict__ b1,
                     const float* __restrict__ W2, const float* __restrict__ b2,
                     const float* __restrict__ W3, const float* __restrict__ b3,
                     float* __restrict__ out) {
  // --- LDS (163,712 B total; 2 workgroups fit in one 320 KB WGP) ---
  __shared__ unsigned int  s_bw1[8 * 4 * 2 * 16 * 8];      // W1comb B-frags [s][n][half][col][j]
  __shared__ unsigned int  s_bw2[2 * 2 * 2 * 16 * 8];      // W2 B-frags
  __shared__ unsigned int  s_abuf[MT_ * 16 * 129];         // bf16 [k | q*k] rows, stride 129 dw
  __shared__ unsigned short s_h1[8 * 16 * 66];             // per-wave h1 tile, stride 66 bf16
  __shared__ float s_q[E_];
  __shared__ float s_bias1[H1_];
  __shared__ float s_scores[208];
  __shared__ float s_red[256];

  const int b    = blockIdx.x;
  const int tid  = threadIdx.x;
  const int lane = tid & 31;
  const int wv   = tid >> 5;       // wave index 0..7
  const int colL = lane & 15;      // N column / A row within tile
  const int lh   = lane >> 4;      // lane half selects K sub-block

  if (tid < E_) s_q[tid] = queries[(size_t)b * E_ + tid];
  __syncthreads();

  // ---- W1comb fragments: rows 0..127 = W1b - W1c (k part), 128..255 = W1d (q*k part)
  for (int i = tid; i < 8192; i += 256) {
    int j = i & 7, col = (i >> 3) & 15, hf = (i >> 7) & 1, n = (i >> 8) & 3, s = i >> 10;
    int k0 = s * 32 + hf * 8 + 2 * j + (j >= 4 ? 8 : 0);
    int h  = n * 16 + col;
    float lo, hi;
    if (k0 < 128) {
      lo = W1[(128 + k0) * H1_ + h] - W1[(256 + k0) * H1_ + h];
      hi = W1[(129 + k0) * H1_ + h] - W1[(257 + k0) * H1_ + h];
    } else {                                   // 384 + (k0-128) = 256 + k0
      lo = W1[(256 + k0) * H1_ + h];
      hi = W1[(257 + k0) * H1_ + h];
    }
    s_bw1[i] = packbf2(lo, hi);
  }
  // ---- W2 fragments
  for (int i = tid; i < 1024; i += 256) {
    int j = i & 7, col = (i >> 3) & 15, hf = (i >> 7) & 1, n2 = (i >> 8) & 1, s2 = i >> 9;
    int k0 = s2 * 32 + hf * 8 + 2 * j + (j >= 4 ? 8 : 0);
    int h  = n2 * 16 + col;
    s_bw2[i] = packbf2(W2[k0 * H2_ + h], W2[(k0 + 1) * H2_ + h]);
  }
  // ---- fold q-part of layer1 into the bias: bias1 = b1 + q @ (W1a + W1c)
  if (tid < H1_) {
    float sum = b1[tid];
    for (int e = 0; e < E_; ++e)
      sum += s_q[e] * (W1[e * H1_ + tid] + W1[(256 + e) * H1_ + tid]);
    s_bias1[tid] = sum;
  }
  __syncthreads();

  const float b2v0 = b2[colL], b2v1 = b2[16 + colL];
  const float w3a = W3[colL], w3b = W3[16 + colL];
  const float b3v = b3[0];

  for (int mt = wv; mt < MT_; mt += 8) {
    unsigned int* ab = &s_abuf[mt * (16 * 129)];
    // build bf16 A tile: dwords 0..63 = keys pairs, 64..127 = (q*k) pairs
    for (int p = lane; p < 16 * 64; p += 32) {
      int r = p >> 6, pi = p & 63;
      int t = mt * 16 + r;
      int e = pi * 2;
      float k0 = 0.f, k1 = 0.f;
      if (t < T_) {
        const float2 kk = *(const float2*)(keys + ((size_t)b * T_ + t) * E_ + e);
        k0 = kk.x; k1 = kk.y;
      }
      ab[r * 129 + pi]      = packbf2(k0, k1);
      ab[r * 129 + 64 + pi] = packbf2(s_q[e] * k0, s_q[e + 1] * k1);
    }

    // ---- layer 1: [16x256] @ [256x64], 8 K-steps x 4 N-tiles
    v8f acc[4];
#pragma unroll
    for (int n = 0; n < 4; ++n) {
      float bv = s_bias1[n * 16 + colL];
#pragma unroll
      for (int j = 0; j < 8; ++j) acc[n][j] = bv;
    }
    const unsigned int* arow = ab + colL * 129;
    // keep rolled: only 1 A-frag + 4 B-frags live per iteration (VGPRs < 256)
#pragma clang loop unroll(disable)
    for (int s = 0; s < 8; ++s) {
      union { v16bf v; unsigned int u[8]; } af;
      int base = s * 16 + lh * 4;
#pragma unroll
      for (int j = 0; j < 4; ++j) { af.u[j] = arow[base + j]; af.u[j + 4] = arow[base + 8 + j]; }
#pragma unroll
      for (int n = 0; n < 4; ++n) {
        union { v16bf v; unsigned int u[8]; } bfr;
        const unsigned int* bp = &s_bw1[(((s * 4 + n) * 2 + lh) * 16 + colL) * 8];
#pragma unroll
        for (int j = 0; j < 8; ++j) bfr.u[j] = bp[j];
        acc[n] = __builtin_amdgcn_wmma_f32_16x16x32_bf16(false, af.v, false, bfr.v,
                                                         (short)0, acc[n], false, false);
      }
    }
    // ReLU -> bf16 -> LDS transpose (D layout -> A layout for layer 2)
    unsigned short* h1w = &s_h1[wv * (16 * 66)];
#pragma unroll
    for (int n = 0; n < 4; ++n)
#pragma unroll
      for (int j = 0; j < 8; ++j) {
        int M = j + 8 * lh;
        float v = acc[n][j]; v = v > 0.f ? v : 0.f;
        h1w[M * 66 + n * 16 + colL] = f2bf(v);
      }

    // ---- layer 2: [16x64] @ [64x32], 2 K-steps x 2 N-tiles
    v8f a2[2];
#pragma unroll
    for (int j = 0; j < 8; ++j) { a2[0][j] = b2v0; a2[1][j] = b2v1; }
    const unsigned int* h1d = (const unsigned int*)h1w;   // stride 33 dwords
#pragma clang loop unroll(disable)
    for (int s2 = 0; s2 < 2; ++s2) {
      union { v16bf v; unsigned int u[8]; } af;
      int rb = colL * 33;
      int base = s2 * 16 + lh * 4;
#pragma unroll
      for (int j = 0; j < 4; ++j) { af.u[j] = h1d[rb + base + j]; af.u[j + 4] = h1d[rb + base + 8 + j]; }
#pragma unroll
      for (int n2 = 0; n2 < 2; ++n2) {
        union { v16bf v; unsigned int u[8]; } bfr;
        const unsigned int* bp = &s_bw2[(((s2 * 2 + n2) * 2 + lh) * 16 + colL) * 8];
#pragma unroll
        for (int j = 0; j < 8; ++j) bfr.u[j] = bp[j];
        a2[n2] = __builtin_amdgcn_wmma_f32_16x16x32_bf16(false, af.v, false, bfr.v,
                                                         (short)0, a2[n2], false, false);
      }
    }

    // ---- layer 3: h2 @ W3 via 16-lane butterfly reduce; apply padding mask
#pragma unroll
    for (int j = 0; j < 8; ++j) {
      float r0 = a2[0][j]; r0 = r0 > 0.f ? r0 : 0.f;
      float r1 = a2[1][j]; r1 = r1 > 0.f ? r1 : 0.f;
      float p = r0 * w3a + r1 * w3b;
#pragma unroll
      for (int m = 1; m < 16; m <<= 1) p += __shfl_xor(p, m, 32);
      if (colL == 0) {
        int t = mt * 16 + j + 8 * lh;
        if (t < T_) {
          float sc = p + b3v;
          if (keys_id[(size_t)b * T_ + t] == 0) sc = NEG_INF_F;
          s_scores[t] = sc;
        }
      }
    }
  }
  __syncthreads();

  // ---- masked softmax over T
  float sv = (tid < T_) ? s_scores[tid] : NEG_INF_F;
  s_red[tid] = sv;
  __syncthreads();
  for (int st = 128; st > 0; st >>= 1) {
    if (tid < st) s_red[tid] = fmaxf(s_red[tid], s_red[tid + st]);
    __syncthreads();
  }
  float mx = s_red[0];
  __syncthreads();
  float ev = (tid < T_) ? __expf(sv - mx) : 0.f;
  s_red[tid] = ev;
  __syncthreads();
  for (int st = 128; st > 0; st >>= 1) {
    if (tid < st) s_red[tid] += s_red[tid + st];
    __syncthreads();
  }
  float ssum = s_red[0];
  __syncthreads();
  if (tid < T_) s_scores[tid] = ev / ssum;   // scores now hold softmax weights
  __syncthreads();

  // ---- weighted mean: out[b,e] = (1/T) * sum_t w[t]*keys[b,t,e]  (keys from LDS bf16 cache)
  int e   = tid & 127;
  int grp = tid >> 7;          // two halves of t-range
  float accv = 0.f;
  int t0 = grp * 100;
  for (int t = t0; t < t0 + 100; ++t) {
    int mt2 = t >> 4, r = t & 15;
    unsigned int d = s_abuf[mt2 * (16 * 129) + r * 129 + (e >> 1)];
    float kv = bf2f((e & 1) ? (d >> 16) : (d & 0xFFFFu));
    accv += s_scores[t] * kv;
  }
  s_red[tid] = accv;
  __syncthreads();
  if (tid < E_)
    out[(size_t)b * E_ + tid] = (s_red[tid] + s_red[tid + 128]) * (1.0f / (float)T_);
}

extern "C" void kernel_launch(void* const* d_in, const int* in_sizes, int n_in,
                              void* d_out, int out_size, void* d_ws, size_t ws_size,
                              hipStream_t stream) {
  const float* queries = (const float*)d_in[0];
  const float* keys    = (const float*)d_in[1];
  const int*   keys_id = (const int*)d_in[2];
  const float* W1 = (const float*)d_in[3];
  const float* b1 = (const float*)d_in[4];
  const float* W2 = (const float*)d_in[5];
  const float* b2 = (const float*)d_in[6];
  const float* W3 = (const float*)d_in[7];
  const float* b3 = (const float*)d_in[8];
  float* out = (float*)d_out;
  hipLaunchKernelGGL(din_attn_kernel, dim3(B_), dim3(256), 0, stream,
                     queries, keys, keys_id, W1, b1, W2, b2, W3, b3, out);
}
